// GATv2_86904368267796
// MI455X (gfx1250) — compile-verified, matrix operand order checked by reading, not run on hardware
//
#include <hip/hip_runtime.h>
#include <hip/hip_bf16.h>

typedef __attribute__((ext_vector_type(16))) _Float16 v16h;
typedef __attribute__((ext_vector_type(8)))  _Float16 v8h;
typedef __attribute__((ext_vector_type(8)))  float    v8f;
typedef __attribute__((ext_vector_type(4)))  float    v4f;

#define DIN 128
#define NEG_SLOPE 0.2f
#define BN_EPS 1e-5f

// ---------------------------------------------------------------------------
// Weight prep: W (K x N, row-major f32) -> Wt (Npad x K, row-major f16),
// rows >= N zero-padded (used to pad the 128x2 head weight to 128x16).
// ---------------------------------------------------------------------------
__global__ void transpose_w_kernel(const float* __restrict__ W,
                                   _Float16* __restrict__ Wt,
                                   int K, int N, int Npad) {
  int idx = blockIdx.x * blockDim.x + threadIdx.x;
  if (idx >= Npad * K) return;
  int n = idx / K, k = idx % K;
  Wt[idx] = (n < N) ? (_Float16)W[(size_t)k * N + n] : (_Float16)0.0f;
}

// ---------------------------------------------------------------------------
// WMMA GEMM: C[M x ldc] = A[M x K=128](f32) * Bt[NT*16 x 128](f16)^T + bias.
// Register-blocked: each wave owns one 16-row M tile and sweeps NT N-tiles,
// holding NT 16x16 f32 accumulators (NT*8 VGPRs). A fragment loaded once per
// K-step (f32 -> f16 cvt in regs), reused across all NT WMMAs.
// blockDim = 32*navesPerBlock; grid.x * waves = M/16.
// A fragment per ISA 16-bit A 16x32 layout; Bt mirrors it by column.
// ---------------------------------------------------------------------------
template <int NT>
__global__ __launch_bounds__(128) void gemm_wmma_kernel(
    const float* __restrict__ A, const _Float16* __restrict__ Bt,
    const float* __restrict__ bias, float* __restrict__ C,
    int ldc, int storeN, int biasN, int relu) {
  int wave = threadIdx.x >> 5;
  int lane = threadIdx.x & 31;
  int nw   = blockDim.x >> 5;
  int tileM = blockIdx.x * nw + wave;
  int r16 = lane & 15;
  int hi  = lane >> 4;  // half-wave: selects K octet pairs

  const float* arow = A + (size_t)(tileM * 16 + r16) * DIN;

  v8f c[NT];
  #pragma unroll
  for (int t = 0; t < NT; ++t) c[t] = (v8f){};

  #pragma unroll
  for (int k0 = 0; k0 < DIN; k0 += 32) {
    int k1 = k0 + hi * 8;        // lanes 0-15: k0..k0+7   | lanes 16-31: +8..+15
    int k2 = k0 + 16 + hi * 8;   // lanes 0-15: k0+16..+23 | lanes 16-31: +24..+31
    v16h a;
    v4f f0 = *(const v4f*)(arow + k1);
    v4f f1 = *(const v4f*)(arow + k1 + 4);
    v4f f2 = *(const v4f*)(arow + k2);
    v4f f3 = *(const v4f*)(arow + k2 + 4);
    #pragma unroll
    for (int i = 0; i < 4; ++i) {
      a[i]      = (_Float16)f0[i];
      a[4 + i]  = (_Float16)f1[i];
      a[8 + i]  = (_Float16)f2[i];
      a[12 + i] = (_Float16)f3[i];
    }
    #pragma unroll
    for (int t = 0; t < NT; ++t) {
      const _Float16* brow = Bt + (size_t)(t * 16 + r16) * DIN;
      v8h b0 = *(const v8h*)(brow + k1);
      v8h b1 = *(const v8h*)(brow + k2);
      v16h b;
      #pragma unroll
      for (int i = 0; i < 8; ++i) { b[i] = b0[i]; b[8 + i] = b1[i]; }
      c[t] = __builtin_amdgcn_wmma_f32_16x16x32_f16(false, a, false, b,
                                                    (short)0, c[t], false, false);
    }
  }

  // C/D layout: VGPR r -> (M=r, lanes 0-15) and (M=r+8, lanes 16-31); N=lane&15
  #pragma unroll
  for (int t = 0; t < NT; ++t) {
    #pragma unroll
    for (int r = 0; r < 8; ++r) {
      int row = tileM * 16 + r + hi * 8;
      int col = t * 16 + (lane & 15);
      float v = c[t][r];
      if (col < biasN) v += bias[col];
      if (relu) v = v > 0.0f ? v : 0.0f;
      if (col < storeN) C[(size_t)row * ldc + col] = v;
    }
  }
}

// ---------------------------------------------------------------------------
// Edge phase. One wave32 per edge, lane -> channel (+32 stride, 4 per lane).
// ---------------------------------------------------------------------------
__device__ __forceinline__ unsigned f32_key(float f) {
  unsigned b = __float_as_uint(f);
  return (b & 0x80000000u) ? ~b : (b | 0x80000000u);
}
__device__ __forceinline__ float key_f32(unsigned k) {
  unsigned b = (k & 0x80000000u) ? (k & 0x7FFFFFFFu) : ~k;
  return __uint_as_float(b);
}

__global__ __launch_bounds__(256) void edge_logit_kernel(
    const int* __restrict__ srcA, const int* __restrict__ dstA, int nE, int nTot,
    const float* __restrict__ xl, const float* __restrict__ xr,
    const float* __restrict__ att,
    float* __restrict__ logit, unsigned* __restrict__ lmaxKey) {
  int wave = threadIdx.x >> 5, lane = threadIdx.x & 31;
  int e = blockIdx.x * 8 + wave;
  if (e >= nTot) return;
  int s = (e < nE) ? srcA[e] : (e - nE);
  int d = (e < nE) ? dstA[e] : (e - nE);
  const float* pl = xl + (size_t)s * DIN;
  const float* pr = xr + (size_t)d * DIN;
  float acc = 0.0f;
  #pragma unroll
  for (int i = 0; i < 4; ++i) {
    int ch = lane + i * 32;
    float v = pl[ch] + pr[ch];
    v = v > 0.0f ? v : v * NEG_SLOPE;   // leaky_relu
    acc += v * att[ch];
  }
  #pragma unroll
  for (int off = 16; off > 0; off >>= 1) acc += __shfl_xor(acc, off, 32);
  if (lane == 0) {
    logit[e] = acc;
    atomicMax(lmaxKey + d, f32_key(acc));
  }
}

__global__ void edge_expsum_kernel(const int* __restrict__ dstA, int nE, int nTot,
                                   const float* __restrict__ logit,
                                   const unsigned* __restrict__ lmaxKey,
                                   float* __restrict__ p, float* __restrict__ denom) {
  int e = blockIdx.x * blockDim.x + threadIdx.x;
  if (e >= nTot) return;
  int d = (e < nE) ? dstA[e] : (e - nE);
  float pe = __expf(logit[e] - key_f32(lmaxKey[d]));
  p[e] = pe;
  __hip_atomic_fetch_add(denom + d, pe, __ATOMIC_RELAXED, __HIP_MEMORY_SCOPE_AGENT);
}

__global__ __launch_bounds__(256) void edge_aggregate_kernel(
    const int* __restrict__ srcA, const int* __restrict__ dstA, int nE, int nTot,
    const float* __restrict__ xl, const float* __restrict__ p,
    const float* __restrict__ denom, float* __restrict__ agg) {
  int wave = threadIdx.x >> 5, lane = threadIdx.x & 31;
  int e = blockIdx.x * 8 + wave;
  if (e >= nTot) return;
  int s = (e < nE) ? srcA[e] : (e - nE);
  int d = (e < nE) ? dstA[e] : (e - nE);
  float alpha = p[e] / denom[d];
  const float* pl = xl + (size_t)s * DIN;
  float* pd = agg + (size_t)d * DIN;
  #pragma unroll
  for (int i = 0; i < 4; ++i) {
    int ch = lane + i * 32;
    __hip_atomic_fetch_add(pd + ch, alpha * pl[ch],
                           __ATOMIC_RELAXED, __HIP_MEMORY_SCOPE_AGENT);
  }
}

// h = relu(bn(agg + bias))   (in place)
__global__ void bn_relu_kernel(float* __restrict__ x, const float* __restrict__ bias,
                               const float* __restrict__ g, const float* __restrict__ be,
                               const float* __restrict__ rm, const float* __restrict__ rv,
                               int total) {
  int i = blockIdx.x * blockDim.x + threadIdx.x;
  if (i >= total) return;
  int ch = i & (DIN - 1);
  float scale = g[ch] * rsqrtf(rv[ch] + BN_EPS);
  float v = (x[i] + bias[ch] - rm[ch]) * scale + be[ch];
  x[i] = v > 0.0f ? v : 0.0f;
}

// ---------------------------------------------------------------------------
extern "C" void kernel_launch(void* const* d_in, const int* in_sizes, int n_in,
                              void* d_out, int out_size, void* d_ws, size_t ws_size,
                              hipStream_t stream) {
  const float* x     = (const float*)d_in[0];
  const int*   ei    = (const int*)  d_in[1];
  const float* Wl0   = (const float*)d_in[2];
  const float* bl0   = (const float*)d_in[3];
  const float* Wr0   = (const float*)d_in[4];
  const float* br0   = (const float*)d_in[5];
  const float* att0  = (const float*)d_in[6];
  const float* bias0 = (const float*)d_in[7];
  const float* g0    = (const float*)d_in[8];
  const float* be0   = (const float*)d_in[9];
  const float* rm0   = (const float*)d_in[10];
  const float* rv0   = (const float*)d_in[11];
  const float* Wl1   = (const float*)d_in[12];
  const float* bl1   = (const float*)d_in[13];
  const float* Wr1   = (const float*)d_in[14];
  const float* br1   = (const float*)d_in[15];
  const float* att1  = (const float*)d_in[16];
  const float* bias1 = (const float*)d_in[17];
  const float* g1    = (const float*)d_in[18];
  const float* be1   = (const float*)d_in[19];
  const float* rm1   = (const float*)d_in[20];
  const float* rv1   = (const float*)d_in[21];
  const float* Wlin1 = (const float*)d_in[22];
  const float* blin1 = (const float*)d_in[23];
  const float* Wlin2 = (const float*)d_in[24];
  const float* blin2 = (const float*)d_in[25];

  const int Nn = in_sizes[0] / DIN;   // 40000
  const int E  = in_sizes[1] / 2;     // 640000
  const int ET = E + Nn;              // + self loops
  const int* srcA = ei;
  const int* dstA = ei + E;

  // workspace carve-up (256B aligned)
  char* ws = (char*)d_ws;
  size_t off = 0;
  auto carve = [&](size_t bytes) -> void* {
    off = (off + 255) & ~(size_t)255;
    void* ptr = ws + off;
    off += bytes;
    return ptr;
  };
  _Float16* wtL0 = (_Float16*)carve(DIN * DIN * 2);
  _Float16* wtR0 = (_Float16*)carve(DIN * DIN * 2);
  _Float16* wtL1 = (_Float16*)carve(DIN * DIN * 2);
  _Float16* wtR1 = (_Float16*)carve(DIN * DIN * 2);
  _Float16* wtM1 = (_Float16*)carve(DIN * DIN * 2);
  _Float16* wtM2 = (_Float16*)carve(16 * DIN * 2);
  float*    xl     = (float*)carve((size_t)Nn * DIN * 4);
  float*    xr     = (float*)carve((size_t)Nn * DIN * 4);
  float*    agg    = (float*)carve((size_t)Nn * DIN * 4);
  float*    logit  = (float*)carve((size_t)ET * 4);
  float*    pbuf   = (float*)carve((size_t)ET * 4);
  unsigned* lmaxK  = (unsigned*)carve((size_t)Nn * 4);
  float*    denom  = (float*)carve((size_t)Nn * 4);

  const int gBlk  = Nn / 64;               // 4 waves/block * 16 rows = 64 rows
  const int hBlk  = Nn / 16;               // head GEMM: 1 wave/block
  const int eBlk  = (ET + 7) / 8;
  const int eThr  = (ET + 255) / 256;
  const int nElem = Nn * DIN;

  // weight prep (f32 -> transposed f16)
  transpose_w_kernel<<<(DIN*DIN + 255) / 256, 256, 0, stream>>>(Wl0,   wtL0, DIN, DIN, DIN);
  transpose_w_kernel<<<(DIN*DIN + 255) / 256, 256, 0, stream>>>(Wr0,   wtR0, DIN, DIN, DIN);
  transpose_w_kernel<<<(DIN*DIN + 255) / 256, 256, 0, stream>>>(Wl1,   wtL1, DIN, DIN, DIN);
  transpose_w_kernel<<<(DIN*DIN + 255) / 256, 256, 0, stream>>>(Wr1,   wtR1, DIN, DIN, DIN);
  transpose_w_kernel<<<(DIN*DIN + 255) / 256, 256, 0, stream>>>(Wlin1, wtM1, DIN, DIN, DIN);
  transpose_w_kernel<<<(16*DIN  + 255) / 256, 256, 0, stream>>>(Wlin2, wtM2, DIN, 2, 16);

  // ---- Layer 0 ----
  gemm_wmma_kernel<8><<<gBlk, 128, 0, stream>>>(x, wtL0, bl0, xl, DIN, DIN, DIN, 0);
  gemm_wmma_kernel<8><<<gBlk, 128, 0, stream>>>(x, wtR0, br0, xr, DIN, DIN, DIN, 0);
  hipMemsetAsync(lmaxK, 0, (size_t)Nn * 4, stream);
  hipMemsetAsync(denom, 0, (size_t)Nn * 4, stream);
  hipMemsetAsync(agg,   0, (size_t)Nn * DIN * 4, stream);
  edge_logit_kernel<<<eBlk, 256, 0, stream>>>(srcA, dstA, E, ET, xl, xr, att0, logit, lmaxK);
  edge_expsum_kernel<<<eThr, 256, 0, stream>>>(dstA, E, ET, logit, lmaxK, pbuf, denom);
  edge_aggregate_kernel<<<eBlk, 256, 0, stream>>>(srcA, dstA, E, ET, xl, pbuf, denom, agg);
  bn_relu_kernel<<<(nElem + 255) / 256, 256, 0, stream>>>(agg, bias0, g0, be0, rm0, rv0, nElem);
  // agg now holds h0

  // ---- Layer 1 ----
  gemm_wmma_kernel<8><<<gBlk, 128, 0, stream>>>(agg, wtL1, bl1, xl, DIN, DIN, DIN, 0);
  gemm_wmma_kernel<8><<<gBlk, 128, 0, stream>>>(agg, wtR1, br1, xr, DIN, DIN, DIN, 0);
  hipMemsetAsync(lmaxK, 0, (size_t)Nn * 4, stream);
  hipMemsetAsync(denom, 0, (size_t)Nn * 4, stream);
  hipMemsetAsync(agg,   0, (size_t)Nn * DIN * 4, stream);   // h0 consumed above
  edge_logit_kernel<<<eBlk, 256, 0, stream>>>(srcA, dstA, E, ET, xl, xr, att1, logit, lmaxK);
  edge_expsum_kernel<<<eThr, 256, 0, stream>>>(dstA, E, ET, logit, lmaxK, pbuf, denom);
  edge_aggregate_kernel<<<eBlk, 256, 0, stream>>>(srcA, dstA, E, ET, xl, pbuf, denom, agg);
  bn_relu_kernel<<<(nElem + 255) / 256, 256, 0, stream>>>(agg, bias1, g1, be1, rm1, rv1, nElem);
  // agg now holds h1

  // ---- MLP head ----
  // hidden = relu(h1 @ Wlin1 + blin1)  (reuse xl buffer)
  gemm_wmma_kernel<8><<<gBlk, 128, 0, stream>>>(agg, wtM1, blin1, xl, DIN, DIN, DIN, 1);
  // out = hidden @ Wlin2 + blin2  (N padded to 16, store 2 cols straight to d_out)
  gemm_wmma_kernel<1><<<hBlk, 32, 0, stream>>>(xl, wtM2, blin2, (float*)d_out, 2, 2, 2, 0);
}
